// LSTM_76304388980760
// MI455X (gfx1250) — compile-verified, matrix operand order checked by reading, not run on hardware
//
#include <hip/hip_runtime.h>

typedef __attribute__((ext_vector_type(16))) __bf16 v16bf;
typedef __attribute__((ext_vector_type(8)))  __bf16 v8bf;
typedef __attribute__((ext_vector_type(8)))  float  v8f;

constexpr int BATCH = 1024;
constexpr int T     = 512;
constexpr int DIN   = 3;
constexpr int H     = 64;
constexpr int ROWS  = 16;   // batch rows per workgroup (WMMA M)
constexpr int SA    = 104;  // row stride (elems) of layer0 A buffer: 96 K + pad (16B-aligned, conflict-free)
constexpr int SH    = 72;   // row stride (elems) of h1 buffer: 64 + pad

__device__ __forceinline__ float fast_rcp(float x) { return __builtin_amdgcn_rcpf(x); }
__device__ __forceinline__ float sigm(float x)   { return fast_rcp(1.0f + __expf(-x)); }
__device__ __forceinline__ float tanh_f(float x) { return 1.0f - 2.0f * fast_rcp(1.0f + __expf(2.0f * x)); }

// A/B operand load from LDS, matching 16-bit WMMA operand layout:
// lanes 0-15 hold K = [kbase..kbase+7] and [kbase+16..kbase+23],
// lanes 16-31 hold K = [kbase+8..kbase+15] and [kbase+24..kbase+31].
__device__ __forceinline__ v16bf ld_a(const __bf16* base, int stride, int m, int lane, int kbase) {
  int half = (lane & 16) ? 8 : 0;
  const __bf16* p = base + m * stride + kbase + half;
  v8bf lo = *(const v8bf*)(p);
  v8bf hi = *(const v8bf*)(p + 16);
  return __builtin_shufflevector(lo, hi, 0,1,2,3,4,5,6,7,8,9,10,11,12,13,14,15);
}

__device__ __forceinline__ v8f wmma_bf(v16bf a, v16bf b, v8f c) {
  return __builtin_amdgcn_wmma_f32_16x16x32_bf16(false, a, false, b, (short)0, c, false, false);
}

// Build a register B-operand from a [256 x 64] row-major f32 weight (B[k][n] = w[n][k]).
__device__ __forceinline__ v16bf mk_b(const float* w, int n, int lane, int kbase) {
  int half = (lane & 16) ? 8 : 0;
  v16bf r;
#pragma unroll
  for (int j = 0; j < 8; ++j) {
    r[j]     = (__bf16)w[n * 64 + kbase + half + j];
    r[8 + j] = (__bf16)w[n * 64 + kbase + 16 + half + j];
  }
  return r;
}

// Extended layer-0 B: K = [w_hh0 (64) | w_ih0 (3) | zeros] up to K=96.
__device__ __forceinline__ v16bf mk_b0(const float* whh0, const float* wih0, int n, int lane, int kbase) {
  int half = (lane & 16) ? 8 : 0;
  v16bf r;
#pragma unroll
  for (int j = 0; j < 8; ++j) {
    int k0 = kbase + half + j;
    int k1 = kbase + 16 + half + j;
    float f0 = (k0 < 64) ? whh0[n * 64 + k0] : ((k0 < 67) ? wih0[n * 3 + (k0 - 64)] : 0.0f);
    float f1 = (k1 < 64) ? whh0[n * 64 + k1] : ((k1 < 67) ? wih0[n * 3 + (k1 - 64)] : 0.0f);
    r[j]     = (__bf16)f0;
    r[8 + j] = (__bf16)f1;
  }
  return r;
}

__global__ __launch_bounds__(128, 1) void lstm_fused(
    const float* __restrict__ x,
    const float* __restrict__ w_ih0, const float* __restrict__ w_hh0,
    const float* __restrict__ b_ih0, const float* __restrict__ b_hh0,
    const float* __restrict__ w_ih1, const float* __restrict__ w_hh1,
    const float* __restrict__ b_ih1, const float* __restrict__ b_hh1,
    const float* __restrict__ w_out, const float* __restrict__ b_out,
    float* __restrict__ out) {

  const int tid  = threadIdx.x;
  const int lane = tid & 31;
  const int wave = tid >> 5;          // 4 waves, wave w owns hidden cols [16w, 16w+16)
  const int lc   = lane & 15;
  const int mAdd = (lane & 16) ? 8 : 0;   // C-tile: VGPR v -> row v (lanes<16) / v+8 (lanes>=16)
  const int colw = wave * 16;
  const int row0 = blockIdx.x * ROWS;

  // ---- double-buffered state tiles in LDS ----
  __shared__ __align__(16) __bf16 HX0[2][ROWS][SA];  // [h0(64) | x(3) | 0-pad] per row
  __shared__ __align__(16) __bf16 H1B[2][ROWS][SH];  // h1(64) + pad

  for (int i = tid; i < 2 * ROWS * SA; i += 128) ((__bf16*)HX0)[i] = (__bf16)0.0f;
  for (int i = tid; i < 2 * ROWS * SH; i += 128) ((__bf16*)H1B)[i] = (__bf16)0.0f;

  // ---- hoist all weights into registers (loop-invariant B operands + biases) ----
  v16bf B0[3][4], Bi1[2][4], Bh1[2][4];
  float bias0r[4], bias1r[4];
#pragma unroll
  for (int g = 0; g < 4; ++g) {               // gate order i,f,g,o
    int n = g * 64 + colw + lc;               // gate-output column this lane produces
    bias0r[g] = b_ih0[n] + b_hh0[n];
    bias1r[g] = b_ih1[n] + b_hh1[n];
#pragma unroll
    for (int kc = 0; kc < 3; ++kc) B0[kc][g]  = mk_b0(w_hh0, w_ih0, n, lane, 32 * kc);
#pragma unroll
    for (int kc = 0; kc < 2; ++kc) Bi1[kc][g] = mk_b(w_ih1, n, lane, 32 * kc);
#pragma unroll
    for (int kc = 0; kc < 2; ++kc) Bh1[kc][g] = mk_b(w_hh1, n, lane, 32 * kc);
  }

  float c0[8], c1[8];
#pragma unroll
  for (int v = 0; v < 8; ++v) { c0[v] = 0.0f; c1[v] = 0.0f; }

  // x staging: threads 0..47 each own (row, k) = (tid/3, tid%3); prefetch distance 2.
  const int xrow = tid / 3, xk = tid - xrow * 3;
  const float* xptr = x + (size_t)(row0 + xrow) * (T * DIN) + xk;

  __syncthreads();  // LDS zero-init complete before x_0 staging touches the same bytes

  float xreg = 0.0f;
  if (tid < 48) {
    HX0[0][xrow][64 + xk] = (__bf16)xptr[0];            // x_0 in place
    xreg = xptr[DIN];                                   // x_1 in flight
  }
  __syncthreads();

  int pb = 0;
  for (int t = 0; t < T; ++t) {
    __bf16*       A0  = &HX0[pb][0][0];       // reads: [h0^{t-1} | x_t]
    __bf16*       A0n = &HX0[pb ^ 1][0][0];   // writes: h0^t (+ x_{t+1})
    const __bf16* H1r = &H1B[pb][0][0];       // reads: h1^{t-1}
    __bf16*       H1w = &H1B[pb ^ 1][0][0];   // writes: h1^t

    // ---------- layer 0: gates = [h0|x] (16x96) @ B0 (96x256); C starts at inline 0 ----------
    v8f acc0[4];
#pragma unroll
    for (int g = 0; g < 4; ++g) acc0[g] = {};
#pragma unroll
    for (int kc = 0; kc < 3; ++kc) {
      v16bf a = ld_a(A0, SA, lc, lane, 32 * kc);
#pragma unroll
      for (int g = 0; g < 4; ++g) acc0[g] = wmma_bf(a, B0[kc][g], acc0[g]);
    }
#pragma unroll
    for (int v = 0; v < 8; ++v) {
      float iv = sigm(acc0[0][v] + bias0r[0]);
      float fv = sigm(acc0[1][v] + bias0r[1]);
      float gv = tanh_f(acc0[2][v] + bias0r[2]);
      float ov = sigm(acc0[3][v] + bias0r[3]);
      c0[v] = fv * c0[v] + iv * gv;
      float hv = ov * tanh_f(c0[v]);
      A0n[(v + mAdd) * SA + colw + lc] = (__bf16)hv;
    }
    if (tid < 48) A0n[xrow * SA + 64 + xk] = (__bf16)xreg;   // stage x_{t+1} alongside h0^t
    __syncthreads();
    if (tid < 48) xreg = (t + 2 < T) ? xptr[(size_t)(t + 2) * DIN] : 0.0f;  // overlap with layer 1

    // ---------- layer 1: gates = h0^t @ w_ih1^T + h1^{t-1} @ w_hh1^T ----------
    v8f acc1[4];
#pragma unroll
    for (int g = 0; g < 4; ++g) acc1[g] = {};
#pragma unroll
    for (int kc = 0; kc < 2; ++kc) {
      v16bf a = ld_a(A0n, SA, lc, lane, 32 * kc);
#pragma unroll
      for (int g = 0; g < 4; ++g) acc1[g] = wmma_bf(a, Bi1[kc][g], acc1[g]);
    }
#pragma unroll
    for (int kc = 0; kc < 2; ++kc) {
      v16bf a = ld_a(H1r, SH, lc, lane, 32 * kc);
#pragma unroll
      for (int g = 0; g < 4; ++g) acc1[g] = wmma_bf(a, Bh1[kc][g], acc1[g]);
    }
#pragma unroll
    for (int v = 0; v < 8; ++v) {
      float iv = sigm(acc1[0][v] + bias1r[0]);
      float fv = sigm(acc1[1][v] + bias1r[1]);
      float gv = tanh_f(acc1[2][v] + bias1r[2]);
      float ov = sigm(acc1[3][v] + bias1r[3]);
      c1[v] = fv * c1[v] + iv * gv;
      float hv = ov * tanh_f(c1[v]);
      H1w[(v + mAdd) * SH + colw + lc] = (__bf16)hv;
    }
    __syncthreads();

    pb ^= 1;
  }

  // ---------- output head: logits = h1_{T-1} @ w_out^T + b_out, softmax ----------
  const __bf16* Hf = &H1B[pb][0][0];  // after even T, pb==0 holds final h1
  if (tid < ROWS) {
    const int r = tid;
    float lg[4];
#pragma unroll
    for (int o = 0; o < 4; ++o) {
      float s = b_out[o];
      for (int k = 0; k < H; ++k) s += (float)Hf[r * SH + k] * w_out[o * H + k];
      lg[o] = s;
    }
    float m = fmaxf(fmaxf(lg[0], lg[1]), fmaxf(lg[2], lg[3]));
    float e[4], sum = 0.0f;
#pragma unroll
    for (int o = 0; o < 4; ++o) { e[o] = __expf(lg[o] - m); sum += e[o]; }
    float rs = fast_rcp(sum);
#pragma unroll
    for (int o = 0; o < 4; ++o) out[(size_t)(row0 + r) * 4 + o] = e[o] * rs;
  }
}

extern "C" void kernel_launch(void* const* d_in, const int* in_sizes, int n_in,
                              void* d_out, int out_size, void* d_ws, size_t ws_size,
                              hipStream_t stream) {
  (void)in_sizes; (void)n_in; (void)out_size; (void)d_ws; (void)ws_size;
  const float* x     = (const float*)d_in[0];
  const float* w_ih0 = (const float*)d_in[1];
  const float* w_hh0 = (const float*)d_in[2];
  const float* b_ih0 = (const float*)d_in[3];
  const float* b_hh0 = (const float*)d_in[4];
  const float* w_ih1 = (const float*)d_in[5];
  const float* w_hh1 = (const float*)d_in[6];
  const float* b_ih1 = (const float*)d_in[7];
  const float* b_hh1 = (const float*)d_in[8];
  const float* w_out = (const float*)d_in[9];
  const float* b_out = (const float*)d_in[10];
  float* out = (float*)d_out;

  lstm_fused<<<dim3(BATCH / ROWS), dim3(128), 0, stream>>>(
      x, w_ih0, w_hh0, b_ih0, b_hh0, w_ih1, w_hh1, b_ih1, b_hh1, w_out, b_out, out);
}
